// GRU_43843026158231
// MI455X (gfx1250) — compile-verified, matrix operand order checked by reading, not run on hardware
//
#include <hip/hip_runtime.h>
#include <hip/hip_bf16.h>
#include <math.h>

// ---------------------------------------------------------------------------
// GRU (B=32, T=2048, D=H=OUT=256) for MI455X gfx1250, wave32 + WMMA f16.
//   K0a: x f32 [B,T,D] -> f16 [T,B,D]   (t-major rows so GEMM tiles == scan tiles)
//   K0b: pack Wg/Wc/Wp f32 -> f16 WMMA-B-fragment layout [tn][tk][lane][16]
//   K1 : Gx = x @ Wg[:D,:] + bg  -> fragment-packed f16 [rowtile][tn][lane][8]
//        Cx = x @ Wc[:D,:] + bc  -> fragment-packed f16
//   K2 : sequential scan, ONE 1024-thread WG (M=32, B-fragment reuse x2),
//        candidate recurrent weights made LDS-resident via TDM
//        (tensor_load_to_lds + s_wait_tensorcnt), gate weights streamed (L2-hot)
//   K3 : out = sigmoid(H @ Wp + bp) -> f32, remapped back to [B,T,OUT]
// ---------------------------------------------------------------------------

typedef __attribute__((ext_vector_type(16))) _Float16 v16h;
typedef __attribute__((ext_vector_type(8)))  _Float16 v8h;
typedef __attribute__((ext_vector_type(8)))  float    v8f;
typedef __attribute__((ext_vector_type(4)))  unsigned int u32x4;
typedef __attribute__((ext_vector_type(8)))  int      i32x8;
typedef __attribute__((ext_vector_type(4)))  int      i32x4;

#define WMMA_F16(a, b, c) \
  __builtin_amdgcn_wmma_f32_16x16x32_f16(false, (a), false, (b), (short)0, (c), false, false)

// ---- A fragment (16x32 f16) from row-major [.. x lda], two 16B loads ------
__device__ __forceinline__ v16h load_frag_a(const _Float16* A, int lda, int row0, int k0) {
  int lane = threadIdx.x & 31;
  int m = lane & 15;
  int kbase = (lane < 16) ? 0 : 8;
  const _Float16* p = A + (size_t)(row0 + m) * lda + k0 + kbase;
  v16h f;
  *(v8h*)&f       = *(const v8h*)p;
  *((v8h*)&f + 1) = *(const v8h*)(p + 16);
  return f;
}

// ---- B fragment from packed layout: one 32B per-lane load -----------------
// P layout: frag(tn, tk) at ((tn*KtStride + tk)*32 + lane)*16 halfs
__device__ __forceinline__ v16h load_frag_bp(const _Float16* P, int KtStride, int tn, int tk) {
  int lane = threadIdx.x & 31;
  return *(const v16h*)(P + (((size_t)tn * KtStride + tk) * 32 + lane) * 16);
}

__device__ __forceinline__ float fast_sigmoid(float x) {
  return 1.0f / (1.0f + __expf(-x));
}

// ---- K0a: x [B,T,D] f32 -> [T,B,D] f16 ------------------------------------
__global__ void cvt_x_tmajor_kernel(const float* __restrict__ x,
                                    _Float16* __restrict__ xt,
                                    int B, int T, int D) {
  size_t total = (size_t)B * T * D;
  size_t stride = (size_t)gridDim.x * blockDim.x;
  for (size_t i = (size_t)blockIdx.x * blockDim.x + threadIdx.x; i < total; i += stride) {
    int d = (int)(i % D);
    size_t rb = i / D;
    int b = (int)(rb % B);
    int t = (int)(rb / B);
    xt[i] = (_Float16)x[((size_t)b * T + t) * D + d];
  }
}

// ---- K0b: pack W [K x N] f32 -> B-fragment f16 layout ---------------------
__global__ void pack_b_kernel(const float* __restrict__ W,
                              _Float16* __restrict__ P, int K, int N) {
  int Kt = K >> 5, Nt = N >> 4;
  size_t total = (size_t)Kt * Nt * 512;
  size_t stride = (size_t)gridDim.x * blockDim.x;
  for (size_t i = (size_t)blockIdx.x * blockDim.x + threadIdx.x; i < total; i += stride) {
    int h    = (int)(i & 15);
    int lane = (int)((i >> 4) & 31);
    size_t tile = i >> 9;
    int tk = (int)(tile % Kt);
    int tn = (int)(tile / Kt);
    int n = lane & 15;
    int kbase = (lane < 16) ? 0 : 8;
    int v = h >> 1, odd = h & 1;
    int k = (v < 4) ? (kbase + 2 * v + odd) : (kbase + 16 + 2 * (v - 4) + odd);
    P[i] = (_Float16)W[(size_t)(tk * 32 + k) * N + tn * 16 + n];
  }
}

// ---- K1: C fragments = A @ Bp + bias, stored fragment-packed f16 ----------
__global__ void __launch_bounds__(256)
gemm_bias_f16p_kernel(const _Float16* __restrict__ A,   // [Rp x K] row-major
                      const _Float16* __restrict__ Bp,  // packed B
                      const float* __restrict__ bias,
                      _Float16* __restrict__ Cp,        // packed C fragments
                      int Nt, int K, int KtStride, int tk0) {
  int wave = threadIdx.x >> 5;
  int rt = blockIdx.x * 8 + wave;
  int row0 = rt * 16;
  int tnb = blockIdx.y * 4;
  v8f acc[4] = {};
  int KT = K >> 5;
  for (int kt = 0; kt < KT; ++kt) {
    v16h a = load_frag_a(A, K, row0, kt * 32);
#pragma unroll
    for (int j = 0; j < 4; ++j) {
      v16h b = load_frag_bp(Bp, KtStride, tnb + j, tk0 + kt);
      acc[j] = WMMA_F16(a, b, acc[j]);
    }
  }
  int lane = threadIdx.x & 31;
  int n = lane & 15;
#pragma unroll
  for (int j = 0; j < 4; ++j) {
    float bv = bias[(tnb + j) * 16 + n];
    v8h o;
#pragma unroll
    for (int v = 0; v < 8; ++v) o[v] = (_Float16)(acc[j][v] + bv);
    *(v8h*)(Cp + ((size_t)rt * Nt + tnb + j) * 256 + lane * 8) = o;
  }
}

// ---- K3: out = sigmoid(A @ Bp + bias), f32, remap t-major -> [B,T,N] ------
__global__ void __launch_bounds__(256)
gemm_bias_sigmoid_kernel(const _Float16* __restrict__ A,   // [T*B x K] t-major
                         const _Float16* __restrict__ Bp,  // packed B
                         const float* __restrict__ bias,
                         float* __restrict__ out,          // [B,T,N]
                         int N, int K, int KtStride, int T) {
  int wave = threadIdx.x >> 5;
  int row0 = (blockIdx.x * 8 + wave) * 16;
  int tnb = blockIdx.y * 4;
  v8f acc[4] = {};
  int KT = K >> 5;
  for (int kt = 0; kt < KT; ++kt) {
    v16h a = load_frag_a(A, K, row0, kt * 32);
#pragma unroll
    for (int j = 0; j < 4; ++j) {
      v16h b = load_frag_bp(Bp, KtStride, tnb + j, kt);
      acc[j] = WMMA_F16(a, b, acc[j]);
    }
  }
  int lane = threadIdx.x & 31;
  int n = lane & 15;
  int mbase = (lane < 16) ? 0 : 8;
#pragma unroll
  for (int j = 0; j < 4; ++j) {
    int cn = tnb * 16 + 16 * j + n;
    float bv = bias[cn];
#pragma unroll
    for (int v = 0; v < 8; ++v) {
      int rp = row0 + mbase + v;     // r' = t*32 + b
      int t = rp >> 5, b = rp & 31;
      out[((size_t)b * T + t) * (size_t)N + cn] = fast_sigmoid(acc[j][v] + bv);
    }
  }
}

// ---- K2: sequential GRU scan, single WG, M=32 -----------------------------
// blockDim = 1024 (32 waves). Gate phase: wave w -> gate col tile w (of 32),
// both M tiles (rows 0-15, 16-31) reuse the same B fragment.
// Candidate phase: wave w -> col tile w&15, M tile w>>4.
// LDS (dynamic, 224 KB): [0,128K) packed cand h-weights (TDM-loaded once),
//   then h16 (16K), rh16 (16K), u f32 (32K), hF f32 (32K).
#define LDS_WC   0
#define LDS_H16  (128 * 1024)
#define LDS_RH16 (144 * 1024)
#define LDS_U    (160 * 1024)
#define LDS_HF   (192 * 1024)
#define LDS_SCAN_BYTES (224 * 1024)

__global__ void __launch_bounds__(1024)
gru_scan_kernel(const _Float16* __restrict__ Gxp,
                const _Float16* __restrict__ Cxp,
                const _Float16* __restrict__ WgP,  // packed [Nt=32][Kt=16]; tk 8..15 = h part
                const _Float16* __restrict__ WcP,  // packed [Nt=16][Kt=16]; tk 8..15 = h part
                _Float16* __restrict__ Hout,       // [T*B, 256] t-major
                int T) {
  extern __shared__ char smem[];
  _Float16* wcL  = (_Float16*)(smem + LDS_WC);   // [tn=16][tk=8][lane][16]
  _Float16* h16  = (_Float16*)(smem + LDS_H16);  // 32 x 256
  _Float16* rh16 = (_Float16*)(smem + LDS_RH16); // 32 x 256
  float*    uLds = (float*)(smem + LDS_U);       // 32 x 256
  float*    hF   = (float*)(smem + LDS_HF);      // 32 x 256

  const int wave = threadIdx.x >> 5;
  const int lane = threadIdx.x & 31;
  const int n = lane & 15;
  const int mbase = (lane < 16) ? 0 : 8;

  for (int i = threadIdx.x; i < 32 * 256; i += blockDim.x) {
    h16[i] = (_Float16)0.0f;
    hF[i] = 0.0f;
  }

#if __has_builtin(__builtin_amdgcn_tensor_load_to_lds)
  // TDM: DMA the candidate h-part weights (16 rows of 8KB, row stride 16KB)
  // into LDS offset 0 in one tensor op. data_size=8B -> 1024 elems/row.
  if (wave == 0) {
    unsigned long long ga = (unsigned long long)(const void*)WcP + 8192ull;  // skip tk 0..7 of tn 0
    u32x4 g0;
    g0[0] = 1u;                                   // count=1 user descriptor
    g0[1] = (unsigned)LDS_WC;                     // lds_addr (bytes)
    g0[2] = (unsigned)(ga & 0xFFFFFFFFu);         // global_addr[31:0]
    g0[3] = (unsigned)((ga >> 32) & 0x1FFFFFFu) | (2u << 30);  // addr[56:32] | type=2
    i32x8 g1;
    g1[0] = 0x00030000;          // workgroup_mask=0, data_size=3 (8B)
    g1[1] = (int)(2048u << 16);  // tensor_dim0 = 2048 (lo16 of field)
    g1[2] = (int)(16u << 16);    // tensor_dim0 hi=0 | tensor_dim1 = 16
    g1[3] = (int)(1024u << 16);  // tensor_dim1 hi=0 | tile_dim0 = 1024
    g1[4] = 16;                  // tile_dim1 = 16, tile_dim2 = 0
    g1[5] = 2048;                // tensor_dim0_stride = 2048 elems (16KB)
    g1[6] = 0;
    g1[7] = 0;
    i32x4 gz4 = {0, 0, 0, 0};
    i32x8 gz8 = {0, 0, 0, 0, 0, 0, 0, 0};
    // 6-arg toolchain variant: (g0, g1, g2, g3, g4, cpol)
    __builtin_amdgcn_tensor_load_to_lds(g0, g1, gz4, gz4, gz8, 0);
    __builtin_amdgcn_s_wait_tensorcnt(0);
  }
#else
  for (int i = threadIdx.x; i < 65536; i += blockDim.x)
    wcL[i] = WcP[(size_t)(i >> 12) * 8192 + 4096 + (i & 4095)];
#endif
  __syncthreads();

  for (int t = 0; t < T; ++t) {
    // ---------- gates: ru = sigmoid(Gx[t] + h @ Wgh), col tile = wave ------
    v8f accg[2];
#pragma unroll
    for (int mt = 0; mt < 2; ++mt) {
      v8h gi = *(const v8h*)(Gxp + (((size_t)(t * 2 + mt) * 32) + wave) * 256 + lane * 8);
#pragma unroll
      for (int v = 0; v < 8; ++v) accg[mt][v] = (float)gi[v];
    }
    if (t + 1 < T) {  // prefetch next step's pre-activations while WMMAs run
      __builtin_prefetch(Gxp + ((size_t)((t + 1) * 2) * 32) * 256 + threadIdx.x * 16, 0, 1);
      __builtin_prefetch(Cxp + ((size_t)((t + 1) * 2) * 16) * 256 + threadIdx.x * 8, 0, 1);
    }
#pragma unroll
    for (int kt = 0; kt < 8; ++kt) {
      v16h b = load_frag_bp(WgP, 16, wave, 8 + kt);  // streamed, reused for 2 M-tiles
#pragma unroll
      for (int mt = 0; mt < 2; ++mt) {
        v16h a = load_frag_a(h16, 256, mt * 16, kt * 32);
        accg[mt] = WMMA_F16(a, b, accg[mt]);
      }
    }
    // waves 0..15 own r (cols 0..255): write r*h f16; waves 16..31 own u
    if (wave < 16) {
      int cg = wave * 16 + n;
#pragma unroll
      for (int mt = 0; mt < 2; ++mt)
#pragma unroll
        for (int v = 0; v < 8; ++v) {
          int m = mt * 16 + mbase + v;
          float g = fast_sigmoid(accg[mt][v]);
          rh16[m * 256 + cg] = (_Float16)(g * hF[m * 256 + cg]);
        }
    } else {
      int cu = (wave - 16) * 16 + n;
#pragma unroll
      for (int mt = 0; mt < 2; ++mt)
#pragma unroll
        for (int v = 0; v < 8; ++v) {
          int m = mt * 16 + mbase + v;
          uLds[m * 256 + cu] = fast_sigmoid(accg[mt][v]);
        }
    }
    __syncthreads();

    // ---------- candidate: c = tanh(Cx[t] + (r*h) @ Wch) -------------------
    int tn = wave & 15;
    int mt = wave >> 4;
    v8f accc;
    {
      v8h ci = *(const v8h*)(Cxp + (((size_t)(t * 2 + mt) * 16) + tn) * 256 + lane * 8);
#pragma unroll
      for (int v = 0; v < 8; ++v) accc[v] = (float)ci[v];
    }
#pragma unroll
    for (int kt = 0; kt < 8; ++kt) {
      v16h a = load_frag_a(rh16, 256, mt * 16, kt * 32);
      v16h b = load_frag_bp(wcL, 8, tn, kt);  // LDS-resident weights (ds_read_b128)
      accc = WMMA_F16(a, b, accc);
    }
    // ---------- h update: wave owns 16 cols x 16 rows ----------------------
    int cc = tn * 16 + n;
#pragma unroll
    for (int v = 0; v < 8; ++v) {
      int m = mt * 16 + mbase + v;
      float c = tanhf(accc[v]);
      float u = uLds[m * 256 + cc];
      float hn = u * hF[m * 256 + cc] + (1.0f - u) * c;
      hF[m * 256 + cc] = hn;
      h16[m * 256 + cc] = (_Float16)hn;
      Hout[((size_t)t * 32 + m) * 256 + cc] = (_Float16)hn;
    }
    __syncthreads();
  }
}

// ---------------------------------------------------------------------------
extern "C" void kernel_launch(void* const* d_in, const int* in_sizes, int n_in,
                              void* d_out, int out_size, void* d_ws, size_t ws_size,
                              hipStream_t stream) {
  const float* x  = (const float*)d_in[0];
  const float* Wg = (const float*)d_in[1];
  const float* bg = (const float*)d_in[2];
  const float* Wc = (const float*)d_in[3];
  const float* bc = (const float*)d_in[4];
  const float* Wp = (const float*)d_in[5];
  const float* bp = (const float*)d_in[6];
  float* out = (float*)d_out;

  const int B = 32, T = 2048, D = 256, H = 256, OUTD = 256;
  const size_t R = (size_t)B * T;  // 65536

  char* ws = (char*)d_ws;
  _Float16* x16  = (_Float16*)ws; ws += R * D * sizeof(_Float16);                  // 32 MB
  _Float16* WgP  = (_Float16*)ws; ws += (size_t)(D + H) * 2 * H * sizeof(_Float16);
  _Float16* WcP  = (_Float16*)ws; ws += (size_t)(D + H) * H * sizeof(_Float16);
  _Float16* WpP  = (_Float16*)ws; ws += (size_t)H * OUTD * sizeof(_Float16);
  _Float16* Gxp  = (_Float16*)ws; ws += R * 2 * H * sizeof(_Float16);              // 64 MB
  _Float16* Cxp  = (_Float16*)ws; ws += R * H * sizeof(_Float16);                  // 32 MB
  _Float16* Ho   = (_Float16*)ws; ws += R * H * sizeof(_Float16);                  // 32 MB
  (void)ws_size; (void)n_in; (void)in_sizes; (void)out_size;

  // K0: transpose-convert x; pack weights into WMMA-B fragment layout
  cvt_x_tmajor_kernel<<<4096, 256, 0, stream>>>(x, x16, B, T, D);
  pack_b_kernel<<<512, 256, 0, stream>>>(Wg, WgP, D + H, 2 * H);  // Kt=16, Nt=32
  pack_b_kernel<<<256, 256, 0, stream>>>(Wc, WcP, D + H, H);      // Kt=16, Nt=16
  pack_b_kernel<<<128, 256, 0, stream>>>(Wp, WpP, H, OUTD);       // Kt=8,  Nt=16

  // K1: time-parallel input projections (bias folded), fragment-packed output
  gemm_bias_f16p_kernel<<<dim3((unsigned)(R / 128), 8), 256, 0, stream>>>(
      x16, WgP, bg, Gxp, /*Nt=*/32, /*K=*/D, /*KtStride=*/16, /*tk0=*/0);
  gemm_bias_f16p_kernel<<<dim3((unsigned)(R / 128), 4), 256, 0, stream>>>(
      x16, WcP, bc, Cxp, /*Nt=*/16, /*K=*/D, /*KtStride=*/16, /*tk0=*/0);

  // K2: sequential scan over T; one WG, M=32, TDM-preloaded candidate weights
  gru_scan_kernel<<<1, 1024, LDS_SCAN_BYTES, stream>>>(Gxp, Cxp, WgP, WcP, Ho, T);

  // K3: out = sigmoid(H @ Wp + bp) -> f32 [B,T,OUT]
  gemm_bias_sigmoid_kernel<<<dim3((unsigned)(R / 128), 4), 256, 0, stream>>>(
      Ho, WpP, bp, out, OUTD, H, /*KtStride=*/8, T);
}
